// GCNConv_65781719105877
// MI455X (gfx1250) — compile-verified, matrix operand order checked by reading, not run on hardware
//
#include <hip/hip_runtime.h>
#include <hip/hip_bf16.h>

typedef __attribute__((ext_vector_type(16))) __bf16 v16bf;
typedef __attribute__((ext_vector_type(8)))  float  v8f;

#define F_DIM 128   // F_IN == UNITS == 128

// ---------------------------------------------------------------------------
// Kernel 1: zero the accumulator (d_out) — atomics will accumulate into it.
// ---------------------------------------------------------------------------
__global__ void gcn_zero_kernel(float4* __restrict__ out, int n4) {
    int i = blockIdx.x * blockDim.x + threadIdx.x;
    if (i < n4) out[i] = make_float4(0.f, 0.f, 0.f, 0.f);
}

// ---------------------------------------------------------------------------
// Kernel 2: H = X @ W using v_wmma_f32_16x16x32_bf16 (f32 accumulate).
// Block = 256 threads (8 waves). Each wave computes a 16x128 output strip,
// so a block covers 128 rows. W (128x128) is converted to bf16 and stored in
// LDS in *fragment-major* order: sW[ktile][ntile][lane][16] so each lane's
// v16bf B-fragment is a contiguous 32-byte LDS read (2x ds_load_b128).
//
// 16-bit A-matrix 16x32 layout (ISA 7.12.2): lane L (half = L>>4, m = L&15),
//   VGPR v in 0..3 : K = 2v   + 8*half + {0,1}
//   VGPR v in 4..7 : K = 16 + 2(v-4) + 8*half + {0,1}
// B (32x16) mirrors this with n = L&15. C: VGPR v -> M = v + 8*half, N = L&15.
// ---------------------------------------------------------------------------
__global__ __launch_bounds__(256) void gcn_gemm_wmma_kernel(
        const float* __restrict__ X,
        const float* __restrict__ W,
        float* __restrict__ H,
        int n_nodes) {
    __shared__ __bf16 sW[4][8][32][16];   // 32 KB

    const int t = threadIdx.x;

    // Cooperative prearranged W load: 1024 lane-fragments, 4 per thread.
    #pragma unroll
    for (int q = 0; q < 4; ++q) {
        const int f     = t * 4 + q;          // 0..1023
        const int ktile = f >> 8;
        const int ntile = (f >> 5) & 7;
        const int lane  = f & 31;
        const int n     = (ntile << 4) + (lane & 15);
        const int half  = lane >> 4;
        #pragma unroll
        for (int v = 0; v < 8; ++v) {
            const int kb = (ktile << 5)
                         + ((v < 4) ? (v * 2) : (16 + (v - 4) * 2))
                         + half * 8;
            sW[ktile][ntile][lane][2 * v + 0] = (__bf16)W[(kb + 0) * F_DIM + n];
            sW[ktile][ntile][lane][2 * v + 1] = (__bf16)W[(kb + 1) * F_DIM + n];
        }
    }
    __syncthreads();

    const int lane    = t & 31;
    const int wave    = t >> 5;
    const int half    = lane >> 4;
    const int rowbase = blockIdx.x * 128 + wave * 16;

    // Clamp (not mask) partial-tile row loads so EXEC stays all-ones for WMMA.
    int m = rowbase + (lane & 15);
    if (m > n_nodes - 1) m = n_nodes - 1;
    const float* __restrict__ xrow = X + (long long)m * F_DIM;

    v8f c[8] = {};   // 8 N-tiles of f32 accumulators (64 VGPRs)

    #pragma unroll
    for (int ktile = 0; ktile < 4; ++ktile) {
        // Load + convert this wave's A fragment (16 bf16 per lane).
        v16bf a;
        #pragma unroll
        for (int v = 0; v < 8; ++v) {
            const int kb = (ktile << 5)
                         + ((v < 4) ? (v * 2) : (16 + (v - 4) * 2))
                         + half * 8;
            const float2 xv = *(const float2*)(xrow + kb);
            a[2 * v + 0] = (__bf16)xv.x;
            a[2 * v + 1] = (__bf16)xv.y;
        }
        // 8 WMMAs reuse the same A against all N-tiles of W.
        #pragma unroll
        for (int nt = 0; nt < 8; ++nt) {
            const v16bf b = *(const v16bf*)(&sW[ktile][nt][lane][0]);
            c[nt] = __builtin_amdgcn_wmma_f32_16x16x32_bf16(
                        /*neg_a=*/false, a, /*neg_b=*/false, b,
                        /*c_mod=*/(short)0, c[nt],
                        /*reuse_a=*/false, /*reuse_b=*/false);
        }
    }

    // Store the 16x128 strip (C layout: M = v + 8*half, N = nt*16 + lane&15).
    const int ncol = lane & 15;
    #pragma unroll
    for (int v = 0; v < 8; ++v) {
        const int mr = rowbase + half * 8 + v;
        if (mr < n_nodes) {
            #pragma unroll
            for (int nt = 0; nt < 8; ++nt)
                H[(long long)mr * F_DIM + nt * 16 + ncol] = c[nt][v];
        }
    }
}

// ---------------------------------------------------------------------------
// Kernel 3: sorted-row SpMM scatter. One wave per contiguous edge chunk;
// lane l owns features [4l, 4l+4). Runs of equal edge_row are accumulated in
// registers and flushed with global_atomic_add_f32 only at run boundaries.
// ---------------------------------------------------------------------------
__global__ void gcn_spmm_kernel(const float* __restrict__ H,
                                const long long* __restrict__ erow,
                                const long long* __restrict__ ecol,
                                const float* __restrict__ eval,
                                float* __restrict__ out,
                                int n_edges, int edges_per_wave) {
    const int wave = (blockIdx.x * blockDim.x + threadIdx.x) >> 5;
    const int lane = threadIdx.x & 31;

    long long e0 = (long long)wave * edges_per_wave;
    if (e0 >= n_edges) return;
    long long e1 = e0 + edges_per_wave;
    if (e1 > n_edges) e1 = n_edges;

    float4 acc = make_float4(0.f, 0.f, 0.f, 0.f);
    long long cur = erow[e0];

    for (long long e = e0; e < e1; ++e) {
        const long long r = erow[e];
        if (r != cur) {
            float* dst = out + cur * F_DIM + lane * 4;
            atomicAdd(dst + 0, acc.x);
            atomicAdd(dst + 1, acc.y);
            atomicAdd(dst + 2, acc.z);
            atomicAdd(dst + 3, acc.w);
            acc = make_float4(0.f, 0.f, 0.f, 0.f);
            cur = r;
        }
        const long long cidx = ecol[e];
        const float     v    = eval[e];
        const float4    hv   = *(const float4*)(H + cidx * F_DIM + lane * 4);
        acc.x = fmaf(v, hv.x, acc.x);
        acc.y = fmaf(v, hv.y, acc.y);
        acc.z = fmaf(v, hv.z, acc.z);
        acc.w = fmaf(v, hv.w, acc.w);
    }
    float* dst = out + cur * F_DIM + lane * 4;
    atomicAdd(dst + 0, acc.x);
    atomicAdd(dst + 1, acc.y);
    atomicAdd(dst + 2, acc.z);
    atomicAdd(dst + 3, acc.w);
}

// ---------------------------------------------------------------------------
// Kernel 4: out = sigmoid(out + bias), vectorized float4, in place.
// ---------------------------------------------------------------------------
__global__ void gcn_finish_kernel(float* __restrict__ out,
                                  const float* __restrict__ bias, int n4) {
    int i = blockIdx.x * blockDim.x + threadIdx.x;
    if (i >= n4) return;
    float4 v = ((const float4*)out)[i];
    const float4 b = *(const float4*)(bias + ((i << 2) & (F_DIM - 1)));
    v.x = 1.f / (1.f + __expf(-(v.x + b.x)));
    v.y = 1.f / (1.f + __expf(-(v.y + b.y)));
    v.z = 1.f / (1.f + __expf(-(v.z + b.z)));
    v.w = 1.f / (1.f + __expf(-(v.w + b.w)));
    ((float4*)out)[i] = v;
}

// ---------------------------------------------------------------------------
// Launch: inputs are X[f32], edge_row[i64], edge_col[i64], edge_val[f32],
// weight[f32], bias[f32]; output f32 [N_NODES, 128]. d_ws holds H (51.2 MB).
// ---------------------------------------------------------------------------
extern "C" void kernel_launch(void* const* d_in, const int* in_sizes, int n_in,
                              void* d_out, int out_size, void* d_ws, size_t ws_size,
                              hipStream_t stream) {
    const float*     X    = (const float*)d_in[0];
    const long long* erow = (const long long*)d_in[1];
    const long long* ecol = (const long long*)d_in[2];
    const float*     eval = (const float*)d_in[3];
    const float*     W    = (const float*)d_in[4];
    const float*     bias = (const float*)d_in[5];
    float*           out  = (float*)d_out;
    float*           H    = (float*)d_ws;

    const int n_nodes = in_sizes[0] / F_DIM;
    const int n_edges = in_sizes[1];
    const int n4      = out_size / 4;

    // 1) zero accumulator
    gcn_zero_kernel<<<(n4 + 255) / 256, 256, 0, stream>>>((float4*)out, n4);

    // 2) H = X @ W  (bf16 WMMA, f32 accumulate); 128 rows per 256-thread block
    gcn_gemm_wmma_kernel<<<(n_nodes + 127) / 128, 256, 0, stream>>>(X, W, H, n_nodes);

    // 3) sorted-run SpMM scatter-add
    const int EPW   = 256;
    const int waves = (n_edges + EPW - 1) / EPW;
    gcn_spmm_kernel<<<(waves * 32 + 255) / 256, 256, 0, stream>>>(
        H, erow, ecol, eval, out, n_edges, EPW);

    // 4) bias + sigmoid
    gcn_finish_kernel<<<(n4 + 255) / 256, 256, 0, stream>>>(out, bias, n4);
}
